// GATLayer_64974265253964
// MI455X (gfx1250) — matrix operand order repaired and checked by reading, not compiled, not run on hardware
//
#include <hip/hip_runtime.h>
#include <hip/hip_bf16.h>

#define N 8192
#define IN_DIM 512
#define OUT_DIM 256

typedef __attribute__((ext_vector_type(16))) __bf16 v16bf;
typedef __attribute__((ext_vector_type(8)))  float  v8f;
typedef __attribute__((ext_vector_type(4)))  float  v4f;

// Pin the WMMA pipeline shape: 8 DS reads (4 B tiles) ahead, then
// interleave 1 WMMA : 2 DS reads, then drain 4 WMMAs.
// Masks: 0x008 = MFMA/WMMA class, 0x100 = DS read.
__device__ __forceinline__ void wmma_pipe_sched() {
    __builtin_amdgcn_sched_group_barrier(0x100, 8, 0);
#pragma unroll
    for (int g = 0; g < 12; ++g) {
        __builtin_amdgcn_sched_group_barrier(0x008, 1, 0);
        __builtin_amdgcn_sched_group_barrier(0x100, 2, 0);
    }
    __builtin_amdgcn_sched_group_barrier(0x008, 4, 0);
}

// ---------------------------------------------------------------------------
// Kernel A: h = x @ fc_w + fc_b   (8192x512x256), bf16 WMMA, f32 accumulate.
// Stores h (f32, row-major) and hbT (bf16, [col][row] transposed for kernel B).
// Workgroup = 8 waves, 128 rows; each wave computes a 16x256 tile.
// ---------------------------------------------------------------------------
__global__ __launch_bounds__(256) void gemm_h_kernel(
    const float* __restrict__ x, const float* __restrict__ fcw,
    const float* __restrict__ fcb, float* __restrict__ h,
    __bf16* __restrict__ hbT)
{
    __shared__ __bf16 ldsW[OUT_DIM * 32];   // [col*32 + k] : W^T slice, bf16

    const int tid  = threadIdx.x;
    const int wave = tid >> 5;
    const int lane = tid & 31;
    const int kh   = lane >> 4;     // half-wave selector
    const int mrow = lane & 15;     // M row of A this lane supplies
    const int i0   = blockIdx.x * 128;
    const int arow = i0 + wave * 16 + mrow;

    v8f zero;
#pragma unroll
    for (int e = 0; e < 8; ++e) zero[e] = 0.0f;
    v8f acc[16];
#pragma unroll
    for (int t = 0; t < 16; ++t) acc[t] = zero;

    for (int k0 = 0; k0 < IN_DIM; k0 += 32) {
        __syncthreads();
        // Stage W[k0..k0+31][0..255] transposed into LDS as bf16 (coalesced in col)
#pragma unroll 4
        for (int k = 0; k < 32; ++k) {
            ldsW[tid * 32 + k] = (__bf16)fcw[(size_t)(k0 + k) * OUT_DIM + tid];
        }
        __syncthreads();

        // A tile (16x32 bf16): lane mrow supplies K = {kh*16..kh*16+15}
        const float* xp = x + (size_t)arow * IN_DIM + k0 + kh * 16;
        const v4f xa0 = *(const v4f*)(xp);
        const v4f xa1 = *(const v4f*)(xp + 4);
        const v4f xa2 = *(const v4f*)(xp + 8);
        const v4f xa3 = *(const v4f*)(xp + 12);
        v16bf a;
#pragma unroll
        for (int e = 0; e < 4; ++e) {
            a[e]      = (__bf16)xa0[e];
            a[4 + e]  = (__bf16)xa1[e];
            a[8 + e]  = (__bf16)xa2[e];
            a[12 + e] = (__bf16)xa3[e];
        }

        __builtin_amdgcn_sched_barrier(0);   // keep A-prep out of WMMA region

        // Depth-4 software-pipelined B-tile loads feeding the WMMA chain.
        v16bf bb[4];
#pragma unroll
        for (int t = 0; t < 4; ++t)
            bb[t] = *(const v16bf*)(ldsW + (t * 16 + mrow) * 32 + kh * 16);
#pragma unroll
        for (int t = 0; t < 16; ++t) {
            const v16bf bcur = bb[t & 3];
            if (t + 4 < 16)
                bb[t & 3] = *(const v16bf*)(ldsW + ((t + 4) * 16 + mrow) * 32 + kh * 16);
            acc[t] = __builtin_amdgcn_wmma_f32_16x16x32_bf16(
                false, a, false, bcur, (short)0, acc[t], false, false);
        }
        wmma_pipe_sched();
    }

    // Epilogue: acc layout -> element (M = v + 8*kh, col = 16t + mrow)
#pragma unroll
    for (int t = 0; t < 16; ++t) {
        const int col  = t * 16 + mrow;
        const float bias = fcb[col];
#pragma unroll
        for (int v = 0; v < 8; ++v) {
            const int r = i0 + wave * 16 + v + 8 * kh;
            const float val = acc[t][v] + bias;
            h[(size_t)r * OUT_DIM + col] = val;
            hbT[(size_t)col * N + r]     = (__bf16)val;
        }
    }
}

// ---------------------------------------------------------------------------
// Kernel C: src[i] = h[i]·a_src + attn_b ; dst[i] = h[i]·a_dst
// One wave (32 lanes) per row; wave32 shfl_xor reduction.
// ---------------------------------------------------------------------------
__global__ __launch_bounds__(256) void proj_kernel(
    const float* __restrict__ h, const float* __restrict__ attw,
    const float* __restrict__ attb, float* __restrict__ srcv,
    float* __restrict__ dstv)
{
    const int wave = threadIdx.x >> 5;
    const int lane = threadIdx.x & 31;
    const int row  = blockIdx.x * 8 + wave;
    const float* hp = h + (size_t)row * OUT_DIM + lane * 8;
    float s = 0.0f, d = 0.0f;
#pragma unroll
    for (int e = 0; e < 8; ++e) {
        const float hv = hp[e];
        s += hv * attw[lane * 8 + e];
        d += hv * attw[OUT_DIM + lane * 8 + e];
    }
#pragma unroll
    for (int m = 16; m >= 1; m >>= 1) {
        s += __shfl_xor(s, m);
        d += __shfl_xor(d, m);
    }
    if (lane == 0) {
        srcv[row] = s + attb[0];
        dstv[row] = d;
    }
}

// ---------------------------------------------------------------------------
// Kernel B: fused GAT attention (flash-style; scores ~N(0,1), no max needed).
//   z_ij = adj_ij * leaky_relu(src_i + dst_j); w = exp(z)
//   out_i = (sum_j w_ij * h_j) / (sum_j w_ij)
// Workgroup = 8 waves x 16 rows = 128 rows; loop j in 32-wide steps.
// adj streamed nontemporally (the 256MB bandwidth floor); hbT tiles staged
// through ping-pong LDS buffers (1 barrier/step, staging loads overlap WMMA),
// B tiles fed to WMMA through a depth-4 register pipeline enforced with
// sched_group_barrier directives.
// ---------------------------------------------------------------------------
__global__ __launch_bounds__(256) void gat_attn_kernel(
    const float* __restrict__ adj, const __bf16* __restrict__ hbT,
    const float* __restrict__ srcv, const float* __restrict__ dstv,
    float* __restrict__ out)
{
    __shared__ __bf16 ldsB[2][OUT_DIM * 32];   // ping-pong h^T tiles, bf16
    __shared__ float  ldsd[2][32];             // ping-pong dst_j tiles

    const int tid  = threadIdx.x;
    const int wave = tid >> 5;
    const int lane = tid & 31;
    const int kh   = lane >> 4;
    const int mrow = lane & 15;
    const int i0   = blockIdx.x * 128;
    const int arow = i0 + wave * 16 + mrow;

    const float s_i = srcv[arow];

    v8f zero;
#pragma unroll
    for (int e = 0; e < 8; ++e) zero[e] = 0.0f;
    v8f acc[16];
#pragma unroll
    for (int t = 0; t < 16; ++t) acc[t] = zero;
    float rsum = 0.0f;

    // Prologue: stage tile j0=0 into buffer 0.
    {
        const uint4* s4 = (const uint4*)(hbT + (size_t)tid * N);
        uint4* d4 = (uint4*)(ldsB[0] + tid * 32);
        d4[0] = s4[0]; d4[1] = s4[1]; d4[2] = s4[2]; d4[3] = s4[3];
        if (tid < 32) ldsd[0][tid] = dstv[tid];
    }

    for (int j0 = 0; j0 < N; j0 += 32) {
        const int cur = (j0 >> 5) & 1;
        const int nxt = cur ^ 1;
        const bool have_next = (j0 + 32) < N;

        // Kick off next tile's staging loads (global -> regs) before the
        // barrier so their latency hides under the other waves' WMMA chains.
        uint4 st0 = {}, st1 = {}, st2 = {}, st3 = {};
        float sd = 0.0f;
        if (have_next) {
            const uint4* s4 = (const uint4*)(hbT + (size_t)tid * N + j0 + 32);
            st0 = s4[0]; st1 = s4[1]; st2 = s4[2]; st3 = s4[3];
            if (tid < 32) sd = dstv[j0 + 32 + tid];
        }

        // adj values for this lane's A-tile slots: K = {kh*16+e}
        const float* ap = adj + (size_t)arow * N + j0 + kh * 16;
        if (j0 + 64 < N) __builtin_prefetch(ap + 64, 0, 0);
        const v4f a0 = __builtin_nontemporal_load((const v4f*)(ap));
        const v4f a1 = __builtin_nontemporal_load((const v4f*)(ap + 4));
        const v4f a2 = __builtin_nontemporal_load((const v4f*)(ap + 8));
        const v4f a3 = __builtin_nontemporal_load((const v4f*)(ap + 12));

        __syncthreads();   // ldsB[cur]/ldsd[cur] ready; prev readers done.

        // Commit next tile immediately: frees st* registers before the WMMA
        // chain (readers of ldsB[nxt] finished before the barrier above;
        // visibility to other waves is ensured by the next loop-top barrier).
        if (have_next) {
            uint4* d4 = (uint4*)(ldsB[nxt] + tid * 32);
            d4[0] = st0; d4[1] = st1; d4[2] = st2; d4[3] = st3;
            if (tid < 32) ldsd[nxt][tid] = sd;
        }

        // Pull dst tile into registers so no stray DS reads sit in the WMMA
        // scheduling region.
        float dv[16];
#pragma unroll
        for (int e = 0; e < 8; ++e) {
            dv[e]     = ldsd[cur][kh * 16 + e];
            dv[8 + e] = ldsd[cur][kh * 16 + 8 + e];
        }

        float av[16];
#pragma unroll
        for (int e = 0; e < 4; ++e) {
            av[e] = a0[e]; av[4 + e] = a1[e]; av[8 + e] = a2[e]; av[12 + e] = a3[e];
        }

        v16bf p;
        float part = 0.0f;
#pragma unroll
        for (int e = 0; e < 16; ++e) {
            const float tv = s_i + dv[e];
            const float lr = tv > 0.0f ? tv : 0.01f * tv;
            const float w  = __expf(av[e] * lr);   // adj==0 -> exp(0)=1 (matches ref)
            part += w;
            p[e] = (__bf16)w;
        }
        rsum += part;

        __builtin_amdgcn_sched_barrier(0);   // isolate WMMA region

        // Depth-4 software-pipelined B-tile loads feeding the WMMA chain.
        const __bf16* bbase = ldsB[cur];
        v16bf bb[4];
#pragma unroll
        for (int t = 0; t < 4; ++t)
            bb[t] = *(const v16bf*)(bbase + (t * 16 + mrow) * 32 + kh * 16);
#pragma unroll
        for (int t = 0; t < 16; ++t) {
            const v16bf bcur = bb[t & 3];
            if (t + 4 < 16)
                bb[t & 3] = *(const v16bf*)(bbase + ((t + 4) * 16 + mrow) * 32 + kh * 16);
            acc[t] = __builtin_amdgcn_wmma_f32_16x16x32_bf16(
                false, p, false, bcur, (short)0, acc[t], false, false);
        }
        wmma_pipe_sched();
    }

    // Row sum: lane l has partial for row M=l%16 over its K half; combine halves.
    const float rs = rsum + __shfl_xor(rsum, 16);
    float denom[8];
#pragma unroll
    for (int v = 0; v < 8; ++v) denom[v] = __shfl(rs, v + 8 * kh);

#pragma unroll
    for (int t = 0; t < 16; ++t) {
        const int col = t * 16 + mrow;
#pragma unroll
        for (int v = 0; v < 8; ++v) {
            const int r = i0 + wave * 16 + v + 8 * kh;
            out[(size_t)r * OUT_DIM + col] = acc[t][v] / denom[v];
        }
    }
}

// ---------------------------------------------------------------------------
extern "C" void kernel_launch(void* const* d_in, const int* in_sizes, int n_in,
                              void* d_out, int out_size, void* d_ws, size_t ws_size,
                              hipStream_t stream) {
    const float* adj  = (const float*)d_in[0];   // (N, N)
    const float* x    = (const float*)d_in[1];   // (N, IN_DIM)
    const float* fcw  = (const float*)d_in[2];   // (IN_DIM, OUT_DIM)
    const float* fcb  = (const float*)d_in[3];   // (OUT_DIM,)
    const float* attw = (const float*)d_in[4];   // (2*OUT_DIM,)
    const float* attb = (const float*)d_in[5];   // scalar
    float* out = (float*)d_out;                  // (N, OUT_DIM) f32

    char* ws = (char*)d_ws;
    float*  h    = (float*)ws;                                  // 8 MB
    __bf16* hbT  = (__bf16*)(ws + (size_t)N * OUT_DIM * 4);     // 4 MB
    float*  srcv = (float*)(ws + (size_t)N * OUT_DIM * 6);      // 32 KB
    float*  dstv = srcv + N;                                    // 32 KB

    gemm_h_kernel<<<N / 128, 256, 0, stream>>>(x, fcw, fcb, h, hbT);
    proj_kernel<<<N / 8, 256, 0, stream>>>(h, attw, attb, srcv, dstv);
    gat_attn_kernel<<<N / 128, 256, 0, stream>>>(adj, hbT, srcv, dstv, out);
}
